// TransformerBlockQuantum_65481071403402
// MI455X (gfx1250) — matrix operand order repaired and checked by reading, not compiled, not run on hardware
//
#include <hip/hip_runtime.h>
#include <hip/hip_fp16.h>
#include <math.h>
#include <stdint.h>

typedef __attribute__((ext_vector_type(16))) _Float16 v16h;
typedef __attribute__((ext_vector_type(8)))  _Float16 v8h;
typedef __attribute__((ext_vector_type(8)))  float    v8f;

#define BB   8
#define TT   2048
#define EE   512
#define HH   8
#define BT   (BB*TT)      // 16384
#define FFND 2048
#define SCALE_ATTN 0.044194173824159216f   // 1/sqrt(512)

// Optional CDNA5 async global->LDS path (compile-safe fallback if absent).
// Probe round showed the builtin takes 4 args; param type is a generic
// pointer to a 16-byte int vector.
#if defined(__has_builtin)
#if __has_builtin(__builtin_amdgcn_global_load_async_to_lds_b128)
#define HAVE_ASYNC_LDS 1
#endif
#endif
#ifdef HAVE_ASYNC_LDS
typedef int v4i_vs __attribute__((vector_size(16)));
#endif

// ---------------------------------------------------------------------------
// WMMA fragment loaders (per CDNA5 ISA 7.12.2 VGPR layouts, wave32)
// A 16x32 f16: lane L -> row m=L&15, kh=L>>4; holds K = kh*8+{0..7}, 16+kh*8+{0..7}
// B 32x16 f16: lane L -> col n=L&15, kh=L>>4; holds K = kh*8+{0..7}, 16+kh*8+{0..7}
// C 16x16 f32: lane L -> col n=L&15; VGPR r holds row m = r + 8*(L>>4)
// ---------------------------------------------------------------------------
__device__ __forceinline__ v16h load_afrag(const _Float16* rowk0, int kh) {
  v8h lo = *(const v8h*)(rowk0 + kh * 8);
  v8h hi = *(const v8h*)(rowk0 + 16 + kh * 8);
  v16h a;
#pragma unroll
  for (int j = 0; j < 8; j++) { a[j] = lo[j]; a[8 + j] = hi[j]; }
  return a;
}
__device__ __forceinline__ v16h load_bfrag(const _Float16* rowk0, int kh) {
  v8h lo = *(const v8h*)(rowk0 + kh * 8);
  v8h hi = *(const v8h*)(rowk0 + 16 + kh * 8);
  v16h b;
#pragma unroll
  for (int j = 0; j < 8; j++) { b[j] = lo[j]; b[8 + j] = hi[j]; }
  return b;
}

// ---------------------------------------------------------------------------
// Casts / gathers
// ---------------------------------------------------------------------------
__global__ void k_cast_f16(const float* __restrict__ src, _Float16* __restrict__ dst, int n) {
  int i = blockIdx.x * blockDim.x + threadIdx.x;
  if (i < n) dst[i] = (_Float16)src[i];
}
__global__ void k_gather_wv(const float* __restrict__ wv, _Float16* __restrict__ wvc) {
  int i = blockIdx.x * blockDim.x + threadIdx.x;
  if (i >= 32 * EE) return;
  int r = i / EE, c = i % EE;
  int srow = (r >> 2) * 64 + (r & 3);
  wvc[i] = (_Float16)wv[srow * EE + c];
}
__global__ void k_gather_wo(const float* __restrict__ wo, _Float16* __restrict__ woc) {
  int i = blockIdx.x * blockDim.x + threadIdx.x;
  if (i >= EE * 32) return;
  int e = i >> 5, c = i & 31;
  woc[i] = (_Float16)wo[e * EE + (c >> 2) * 64 + (c & 3)];
}

// ---------------------------------------------------------------------------
// Register-blocked WMMA GEMM: each wave computes a (16*NTM) x (16*NTN) macro
// tile of C = A[M,K] * W[N,K]^T. A frags reused across NTN, B across NTM:
// (NTM+NTN) KB of loads feed NTM*NTN WMMAs per K-step (~21 flop/B at 2x4).
// ---------------------------------------------------------------------------
template <int NTM, int NTN>
__global__ __launch_bounds__(256) void k_gemm_blk(
    const _Float16* __restrict__ A, const _Float16* __restrict__ W,
    float* __restrict__ Cf, _Float16* __restrict__ Ch, int M, int N, int K) {
  int lane = threadIdx.x & 31;
  int wave = threadIdx.x >> 5;
  int tid = blockIdx.x * 8 + wave;
  int ntn = N / (16 * NTN);
  int tN = (tid % ntn) * NTN;   // units of 16-col tiles
  int tM = (tid / ntn) * NTM;   // units of 16-row tiles
  if (tM >= (M >> 4)) return;
  int m = lane & 15, kh = lane >> 4;
  const _Float16* arow[NTM];
  const _Float16* brow[NTN];
#pragma unroll
  for (int i = 0; i < NTM; i++) arow[i] = A + (size_t)((tM + i) * 16 + m) * K;
#pragma unroll
  for (int j = 0; j < NTN; j++) brow[j] = W + (size_t)((tN + j) * 16 + m) * K;
  v8f c[NTM][NTN];
#pragma unroll
  for (int i = 0; i < NTM; i++)
#pragma unroll
    for (int j = 0; j < NTN; j++) c[i][j] = (v8f){};

  for (int k0 = 0; k0 < K; k0 += 32) {
    v16h a[NTM], b[NTN];
#pragma unroll
    for (int i = 0; i < NTM; i++) a[i] = load_afrag(arow[i] + k0, kh);
#pragma unroll
    for (int j = 0; j < NTN; j++) b[j] = load_bfrag(brow[j] + k0, kh);
#pragma unroll
    for (int i = 0; i < NTM; i++)
#pragma unroll
      for (int j = 0; j < NTN; j++)
        c[i][j] = __builtin_amdgcn_wmma_f32_16x16x32_f16(false, a[i], false, b[j],
                                                         (short)0, c[i][j], false, false);
  }
#pragma unroll
  for (int i = 0; i < NTM; i++)
#pragma unroll
    for (int j = 0; j < NTN; j++) {
      int col = (tN + j) * 16 + m;
#pragma unroll
      for (int r = 0; r < 8; r++) {
        size_t row = (size_t)((tM + i) * 16 + r + 8 * kh);
        if (Cf) Cf[row * N + col] = c[i][j][r];
        else    Ch[row * N + col] = (_Float16)c[i][j][r];
      }
    }
}

// ---------------------------------------------------------------------------
// Flash attention: softmax(Q K^T / sqrt(E)) * V32 with online softmax.
// 8 waves/block share the key loop: K tiles staged cooperatively into LDS
// (async load-to-LDS when available), V staged transposed. One wave = 16 rows.
// ---------------------------------------------------------------------------
#define KSTR 520   // padded K-tile row stride (halves): 16-lane frag reads conflict-free
#define VSTR 40    // padded Vt row stride (halves)
__global__ __launch_bounds__(256) void k_flash_attn(
    const _Float16* __restrict__ Q, const _Float16* __restrict__ Km,
    const _Float16* __restrict__ V32, float* __restrict__ O32) {
  __shared__ _Float16 kt_lds[32][KSTR];    // 32 key rows x 512 features
  __shared__ _Float16 vt_lds[32][VSTR];    // transposed V tile: [channel][key]
  __shared__ _Float16 plds[8][16][32];     // per-wave P round-trip (C->A layout)
  int tid  = threadIdx.x;
  int lane = tid & 31;
  int wave = tid >> 5;
  int b = blockIdx.y;
  int rowbase = blockIdx.x * 128 + wave * 16;
  int m = lane & 15, kh = lane >> 4;
  const _Float16* qrow = Q + (size_t)(b * TT + rowbase + m) * EE;
  const _Float16* Kb = Km + (size_t)b * TT * EE;
  const _Float16* Vb = V32 + (size_t)b * TT * 32;
  v8f o1 = {}, o2 = {};
  float mr[8], lr[8];
#pragma unroll
  for (int r = 0; r < 8; r++) { mr[r] = -3.0e38f; lr[r] = 0.f; }

  for (int kt0 = 0; kt0 < TT; kt0 += 32) {
    __syncthreads();  // previous iteration's consumers done with kt_lds/vt_lds
    // ---- stage K tile: 32 rows x 1KB = 2048 x 16B chunks, 8 per thread ----
#pragma unroll
    for (int i = 0; i < 8; i++) {
      int ch = tid + 256 * i;
      int row = ch >> 6;            // 64 chunks per row
      int col = (ch & 63) * 8;      // halves
      const _Float16* src = Kb + (size_t)(kt0 + row) * EE + col;
      _Float16* dst = &kt_lds[row][col];
#ifdef HAVE_ASYNC_LDS
      __builtin_amdgcn_global_load_async_to_lds_b128((v4i_vs*)(void*)src,
                                                     (v4i_vs*)(void*)dst, 0, 0);
#else
      *(uint4*)dst = *(const uint4*)src;
#endif
    }
    // ---- stage V tile transposed: 32 keys x 32 channels ----
#pragma unroll
    for (int i = 0; i < 4; i++) {
      int e = tid + 256 * i;
      int krow = e >> 5, ccol = e & 31;
      vt_lds[ccol][krow] = Vb[(size_t)(kt0 + krow) * 32 + ccol];
    }
#ifdef HAVE_ASYNC_LDS
#if __has_builtin(__builtin_amdgcn_s_wait_asynccnt)
    __builtin_amdgcn_s_wait_asynccnt(0);
#else
    asm volatile("s_wait_asynccnt 0x0" ::: "memory");
#endif
#endif
    __syncthreads();

    // ---- S = Q K^T over 512 features, two 16-key sub-tiles ----
    v8f sA = {}, sB = {};
    const _Float16* kArow = &kt_lds[m][0];
    const _Float16* kBrow = &kt_lds[16 + m][0];
    for (int k0 = 0; k0 < EE; k0 += 32) {
      v16h a  = load_afrag(qrow + k0, kh);
      v16h bA = load_bfrag(kArow + k0, kh);
      v16h bB = load_bfrag(kBrow + k0, kh);
      sA = __builtin_amdgcn_wmma_f32_16x16x32_f16(false, a, false, bA, (short)0, sA, false, false);
      sB = __builtin_amdgcn_wmma_f32_16x16x32_f16(false, a, false, bB, (short)0, sB, false, false);
    }
    // ---- online softmax per accumulator row r (global row = r + 8*kh) ----
#pragma unroll
    for (int r = 0; r < 8; r++) {
      float va = sA[r] * SCALE_ATTN, vb = sB[r] * SCALE_ATTN;
      float mx = fmaxf(va, vb);
#pragma unroll
      for (int off = 1; off < 16; off <<= 1) mx = fmaxf(mx, __shfl_xor(mx, off, 32));
      float mn = fmaxf(mr[r], mx);
      float alpha = __expf(mr[r] - mn);
      float pa = __expf(va - mn), pb = __expf(vb - mn);
      float ps = pa + pb;
#pragma unroll
      for (int off = 1; off < 16; off <<= 1) ps += __shfl_xor(ps, off, 32);
      lr[r] = lr[r] * alpha + ps;
      mr[r] = mn;
      o1[r] *= alpha; o2[r] *= alpha;
      int row = r + 8 * kh;
      plds[wave][row][m]      = (_Float16)pa;
      plds[wave][row][16 + m] = (_Float16)pb;
    }
    asm volatile("s_wait_dscnt 0x0" ::: "memory");  // wave-local C->A layout round trip
    v16h pf;
#pragma unroll
    for (int j = 0; j < 8; j++) {
      pf[j]     = plds[wave][m][kh * 8 + j];
      pf[8 + j] = plds[wave][m][16 + kh * 8 + j];
    }
    // ---- V b-frags from transposed LDS tile: contiguous 16B reads ----
    v16h bv1 = load_bfrag(&vt_lds[m][0], kh);
    v16h bv2 = load_bfrag(&vt_lds[16 + m][0], kh);
    o1 = __builtin_amdgcn_wmma_f32_16x16x32_f16(false, pf, false, bv1, (short)0, o1, false, false);
    o2 = __builtin_amdgcn_wmma_f32_16x16x32_f16(false, pf, false, bv2, (short)0, o2, false, false);
  }
#pragma unroll
  for (int r = 0; r < 8; r++) {
    size_t row = (size_t)(b * TT + rowbase + r + 8 * kh);
    float inv = 1.0f / lr[r];
    O32[row * 32 + m]      = o1[r] * inv;
    O32[row * 32 + 16 + m] = o2[r] * inv;
  }
}

// ---------------------------------------------------------------------------
// 4-qubit head circuit: one thread per (h, bt). 16-amplitude state in regs.
// ---------------------------------------------------------------------------
__global__ void k_head_circuit(const float* __restrict__ O32,
                               const float* __restrict__ hp,   // [H][2][4]
                               _Float16* __restrict__ qoutc) { // [BT][32]
  int id = blockIdx.x * blockDim.x + threadIdx.x;
  if (id >= HH * BT) return;
  int h = id >> 14, bt = id & (BT - 1);
  float st[16];
#pragma unroll
  for (int i = 0; i < 16; i++) st[i] = 0.f;
  st[0] = 1.f;

  auto ry = [&](int q, float th) {
    float c = cosf(0.5f * th), s = sinf(0.5f * th);
    int bit = 1 << (3 - q);
#pragma unroll
    for (int i = 0; i < 16; i++) {
      if ((i & bit) == 0) {
        float s0 = st[i], s1 = st[i | bit];
        st[i]       = c * s0 - s * s1;
        st[i | bit] = s * s0 + c * s1;
      }
    }
  };
  auto cnot = [&](int ctrl, int tgt) {
    int pc = 1 << (3 - ctrl), pt = 1 << (3 - tgt);
#pragma unroll
    for (int i = 0; i < 16; i++) {
      if ((i & pc) && !(i & pt)) {
        float t = st[i]; st[i] = st[i | pt]; st[i | pt] = t;
      }
    }
  };
#pragma unroll
  for (int l = 0; l < 2; l++) {
#pragma unroll
    for (int q = 0; q < 4; q++) ry(q, hp[(h * 2 + l) * 4 + q]);
    cnot(0, 1); cnot(1, 2); cnot(2, 3); cnot(3, 0);
  }
  const float* xrow = O32 + (size_t)bt * 32 + h * 4;
#pragma unroll
  for (int q = 0; q < 4; q++) ry(q, xrow[q]);
#pragma unroll
  for (int q = 0; q < 4; q++) {
    int p = 3 - q; float z = 0.f;
#pragma unroll
    for (int i = 0; i < 16; i++) {
      float pr = st[i] * st[i];
      z += ((i >> p) & 1) ? -pr : pr;
    }
    qoutc[(size_t)bt * 32 + h * 4 + q] = (_Float16)z;
  }
}

// ---------------------------------------------------------------------------
// 8-qubit FFN circuit: one block (256 thr) per element; state in LDS.
// ---------------------------------------------------------------------------
__global__ __launch_bounds__(256) void k_ffn_circuit(
    const float* __restrict__ X1, const float* __restrict__ fp,
    float* __restrict__ FQ) {
  __shared__ float st[256];
  __shared__ float zq[8];
  int tid = threadIdx.x;
  int bt  = blockIdx.x;
  st[tid] = (tid == 0) ? 1.f : 0.f;
  if (tid < 8) zq[tid] = 0.f;
  __syncthreads();

  for (int q = 0; q < 8; q++) {   // RY layer (shared params)
    float th = 0.5f * fp[q];
    float c = cosf(th), s = sinf(th);
    int mask = 1 << (7 - q);
    float own = st[tid], other = st[tid ^ mask];
    __syncthreads();
    st[tid] = (tid & mask) ? (s * other + c * own) : (c * own - s * other);
    __syncthreads();
  }
  for (int q = 0; q < 7; q++) {   // CNOT chain
    int pc = 1 << (7 - q), pt = 1 << (7 - (q + 1));
    int src = (tid & pc) ? (tid ^ pt) : tid;
    float v = st[src];
    __syncthreads();
    st[tid] = v;
    __syncthreads();
  }
  {  // cnot(7, 0)
    int pc = 1 << 0, pt = 1 << 7;
    int src = (tid & pc) ? (tid ^ pt) : tid;
    float v = st[src];
    __syncthreads();
    st[tid] = v;
    __syncthreads();
  }
  const float* xrow = X1 + (size_t)bt * EE;   // data RY layer
  for (int q = 0; q < 8; q++) {
    float th = 0.5f * xrow[q];
    float c = cosf(th), s = sinf(th);
    int mask = 1 << (7 - q);
    float own = st[tid], other = st[tid ^ mask];
    __syncthreads();
    st[tid] = (tid & mask) ? (s * other + c * own) : (c * own - s * other);
    __syncthreads();
  }
  float amp = st[tid];
  float pr = amp * amp;
#pragma unroll
  for (int q = 0; q < 8; q++) {
    float contrib = ((tid >> (7 - q)) & 1) ? -pr : pr;
    atomicAdd(&zq[q], contrib);   // ds_add_f32
  }
  __syncthreads();
  if (tid < 8) FQ[(size_t)bt * 8 + tid] = zq[tid];
}

// ---------------------------------------------------------------------------
__global__ void k_ffn_fc1(const float* __restrict__ FQ, const float* __restrict__ W1,
                          _Float16* __restrict__ H16n) {
  int i = blockIdx.x * blockDim.x + threadIdx.x;
  if (i >= BT * FFND) return;
  int bt = i >> 11, j = i & (FFND - 1);
  const float* f = FQ + (size_t)bt * 8;
  const float* w = W1 + (size_t)j * 8;
  float acc = 0.f;
#pragma unroll
  for (int q = 0; q < 8; q++) acc += f[q] * w[q];
  H16n[i] = (_Float16)fmaxf(acc, 0.f);
}

// ---------------------------------------------------------------------------
__global__ __launch_bounds__(256) void k_add_ln(
    const float* __restrict__ A, const float* __restrict__ Bv,
    const float* __restrict__ g, const float* __restrict__ be,
    float* __restrict__ Out) {
  int lane = threadIdx.x & 31;
  int wave = threadIdx.x >> 5;
  int row = blockIdx.x * 8 + wave;
  const float* a = A + (size_t)row * EE;
  const float* b = Bv + (size_t)row * EE;
  float v[16];
  float s = 0.f;
#pragma unroll
  for (int i = 0; i < 16; i++) { v[i] = a[lane + 32 * i] + b[lane + 32 * i]; s += v[i]; }
#pragma unroll
  for (int off = 1; off < 32; off <<= 1) s += __shfl_xor(s, off, 32);
  float mean = s * (1.0f / EE);
  float var = 0.f;
#pragma unroll
  for (int i = 0; i < 16; i++) { float d = v[i] - mean; var += d * d; }
#pragma unroll
  for (int off = 1; off < 32; off <<= 1) var += __shfl_xor(var, off, 32);
  float inv = rsqrtf(var * (1.0f / EE) + 1e-5f);
  float* o = Out + (size_t)row * EE;
#pragma unroll
  for (int i = 0; i < 16; i++) {
    int c = lane + 32 * i;
    o[c] = (v[i] - mean) * inv * g[c] + be[c];
  }
}

// ---------------------------------------------------------------------------
extern "C" void kernel_launch(void* const* d_in, const int* in_sizes, int n_in,
                              void* d_out, int out_size, void* d_ws, size_t ws_size,
                              hipStream_t stream) {
  const float* x   = (const float*)d_in[0];
  const float* wq  = (const float*)d_in[1];
  const float* wk  = (const float*)d_in[2];
  const float* wv  = (const float*)d_in[3];
  const float* wo  = (const float*)d_in[4];
  const float* hp  = (const float*)d_in[5];
  const float* fp  = (const float*)d_in[6];
  const float* w1  = (const float*)d_in[7];
  const float* w2  = (const float*)d_in[8];
  const float* g1  = (const float*)d_in[9];
  const float* b1  = (const float*)d_in[10];
  const float* g2  = (const float*)d_in[11];
  const float* b2  = (const float*)d_in[12];
  float* out = (float*)d_out;

  char* ws = (char*)d_ws;
  size_t off = 0;
  auto salloc = [&](size_t bytes) -> void* {
    void* p = ws + off;
    off += (bytes + 255) & ~(size_t)255;
    return p;
  };
  _Float16* X16   = (_Float16*)salloc((size_t)BT * EE * 2);
  _Float16* WQ16  = (_Float16*)salloc((size_t)EE * EE * 2);
  _Float16* WK16  = (_Float16*)salloc((size_t)EE * EE * 2);
  _Float16* WVC16 = (_Float16*)salloc((size_t)32 * EE * 2);
  _Float16* WOC16 = (_Float16*)salloc((size_t)EE * 32 * 2);
  _Float16* W216  = (_Float16*)salloc((size_t)EE * FFND * 2);
  _Float16* Q16   = (_Float16*)salloc((size_t)BT * EE * 2);
  _Float16* K16   = (_Float16*)salloc((size_t)BT * EE * 2);
  _Float16* V32   = (_Float16*)salloc((size_t)BT * 32 * 2);
  float*    O32   = (float*)   salloc((size_t)BT * 32 * 4);
  _Float16* QOUTC = (_Float16*)salloc((size_t)BT * 32 * 2);
  float*    ATTN  = (float*)   salloc((size_t)BT * EE * 4);
  float*    X1    = (float*)   salloc((size_t)BT * EE * 4);
  float*    FQ    = (float*)   salloc((size_t)BT * 8 * 4);
  _Float16* H16n  = (_Float16*)salloc((size_t)BT * FFND * 2);
  float*    FOUT  = (float*)   salloc((size_t)BT * EE * 4);

  // --- precision conversion / gathers ---
  k_cast_f16<<<(BT * EE + 255) / 256, 256, 0, stream>>>(x, X16, BT * EE);
  k_cast_f16<<<(EE * EE + 255) / 256, 256, 0, stream>>>(wq, WQ16, EE * EE);
  k_cast_f16<<<(EE * EE + 255) / 256, 256, 0, stream>>>(wk, WK16, EE * EE);
  k_gather_wv<<<(32 * EE + 255) / 256, 256, 0, stream>>>(wv, WVC16);
  k_gather_wo<<<(EE * 32 + 255) / 256, 256, 0, stream>>>(wo, WOC16);
  k_cast_f16<<<(EE * FFND + 255) / 256, 256, 0, stream>>>(w2, W216, EE * FFND);

  // --- Q, K projections (32x64 macro tiles); V restricted to 32 columns ---
  k_gemm_blk<2, 4><<<(BT / 32) * (EE / 64) / 8, 256, 0, stream>>>(X16, WQ16, nullptr, Q16, BT, EE, EE);
  k_gemm_blk<2, 4><<<(BT / 32) * (EE / 64) / 8, 256, 0, stream>>>(X16, WK16, nullptr, K16, BT, EE, EE);
  k_gemm_blk<2, 2><<<(BT / 32) * (32 / 32) / 8, 256, 0, stream>>>(X16, WVC16, nullptr, V32, BT, 32, EE);

  // --- attention with online softmax, LDS-staged K/V ---
  k_flash_attn<<<dim3(TT / 128, BB), 256, 0, stream>>>(Q16, K16, V32, O32);

  // --- per-head 4-qubit circuits ---
  k_head_circuit<<<(HH * BT + 255) / 256, 256, 0, stream>>>(O32, hp, QOUTC);

  // --- attn_out = qout @ wo^T (compacted to K=32) ---
  k_gemm_blk<2, 4><<<(BT / 32) * (EE / 64) / 8, 256, 0, stream>>>(QOUTC, WOC16, ATTN, nullptr, BT, EE, 32);

  // --- x1 = LN(x + attn_out) ---
  k_add_ln<<<BT / 8, 256, 0, stream>>>(x, ATTN, g1, b1, X1);

  // --- 8-qubit FFN circuit ---
  k_ffn_circuit<<<BT, 256, 0, stream>>>(X1, fp, FQ);

  // --- h = relu(f @ w1^T), then y = h @ w2^T ---
  k_ffn_fc1<<<(BT * FFND + 255) / 256, 256, 0, stream>>>(FQ, w1, H16n);
  k_gemm_blk<2, 4><<<(BT / 32) * (EE / 64) / 8, 256, 0, stream>>>(H16n, W216, FOUT, nullptr, BT, EE, FFND);

  // --- out = LN(x1 + ffn) ---
  k_add_ln<<<BT / 8, 256, 0, stream>>>(X1, FOUT, g2, b2, out);
}
